// GaussianMapLayer_39900246180167
// MI455X (gfx1250) — matrix operand order repaired and checked by reading, not compiled, not run on hardware
//
#include <hip/hip_runtime.h>

// Problem constants (match reference)
#define B_N   4096
#define A_N   8
#define DI_N  64
#define DS_N  64
#define C_N   32
#define H_N   256
#define MW_N  1024
#define MH_N  1024
#define FOURH 1024

typedef __attribute__((ext_vector_type(16))) __bf16 v16bf;
typedef __attribute__((ext_vector_type(8)))  float  v8f;

// workspace layout (bf16 elements)
// others weights: 8 agents x 10 ksteps x 64 ntiles x 1024 (512 hi + 512 lo)
// self weights  : 11 ksteps x 64 ntiles x 1024
// otm weights   : 8 ksteps x 2 ntiles x 1024
static const int AG_STRIDE = 10 * 64 * 1024;          // 655360
static const int OFF_SELF  = A_N * AG_STRIDE;         // 5242880
static const int OFF_OTM   = OFF_SELF + 11 * 64 * 1024; // 5963776

__device__ __forceinline__ v8f wmma_bf16(v16bf a, v16bf b, v8f c) {
  return __builtin_amdgcn_wmma_f32_16x16x32_bf16(false, a, false, b, (short)0, c,
                                                 false, false);
}

__device__ __forceinline__ float sigmoidf_(float x) {
  return 1.0f / (1.0f + __expf(-x));
}

// Build A-fragment (16x32 bf16, ISA A-layout) + lo residual from f32 matrix.
// Lane L: half = L>>4, row = row0 + (L&15).
// slots 0..7  -> K = k0 + 8*half + s
// slots 8..15 -> K = k0 + 16 + 8*half + (s-8)
__device__ __forceinline__ void load_a_frag(const float* base, int ld, int row0,
                                            int k0, v16bf& hi, v16bf& lo) {
  int lane = threadIdx.x & 31;
  int half = lane >> 4;
  int m    = lane & 15;
  const float* p = base + (size_t)(row0 + m) * ld + k0 + 8 * half;
  float vals[16];
  *(float4*)(vals + 0)  = *(const float4*)(p);
  *(float4*)(vals + 4)  = *(const float4*)(p + 4);
  *(float4*)(vals + 8)  = *(const float4*)(p + 16);
  *(float4*)(vals + 12) = *(const float4*)(p + 20);
#pragma unroll
  for (int s = 0; s < 16; ++s) {
    float v  = vals[s];
    __bf16 h = (__bf16)v;
    hi[s]    = h;
    lo[s]    = (__bf16)(v - (float)h);
  }
}

// A-fragment for comm columns: rows gathered through pos_self -> map2.
__device__ __forceinline__ void load_a_frag_comm(const float* map, const int* pos_s,
                                                 int row0, v16bf& hi, v16bf& lo) {
  int lane = threadIdx.x & 31;
  int half = lane >> 4;
  int m    = lane & 15;
  int r    = row0 + m;
  int flat = pos_s[2 * r] * MH_N + pos_s[2 * r + 1];
  const float* p = map + (size_t)flat * C_N + 8 * half;
  float vals[16];
  *(float4*)(vals + 0)  = *(const float4*)(p);
  *(float4*)(vals + 4)  = *(const float4*)(p + 4);
  *(float4*)(vals + 8)  = *(const float4*)(p + 16);
  *(float4*)(vals + 12) = *(const float4*)(p + 20);
#pragma unroll
  for (int s = 0; s < 16; ++s) {
    float v  = vals[s];
    __bf16 h = (__bf16)v;
    hi[s]    = h;
    lo[s]    = (__bf16)(v - (float)h);
  }
}

// ---------------- kernel 0: pack weights into WMMA-B fragment layout ----------
// B (KxN) 16-bit layout: lane = (n%16) + 16*((kk>>3)&1), slot = (kk&7) + 8*(kk>>4) per 32-K step.
__global__ void swizzle_weights(const float* __restrict__ Wk_self,
                                const float* __restrict__ Wr_self,
                                const float* __restrict__ Wk_o,
                                const float* __restrict__ Wr_o,
                                const float* __restrict__ W_otm,
                                __bf16* __restrict__ ws) {
  const int N_OTH  = A_N * 320 * FOURH;   // 2,621,440
  const int N_SELF = 352 * FOURH;         //   360,448
  const int N_OTM  = 256 * C_N;           //     8,192
  int idx = blockIdx.x * blockDim.x + threadIdx.x;

  float v;
  int k, n, NT;
  __bf16* base;
  if (idx < N_OTH) {
    int a   = idx / (320 * FOURH);
    int rem = idx % (320 * FOURH);
    k = rem / FOURH;
    n = rem % FOURH;
    v = (k < DI_N) ? Wk_o[((size_t)a * DI_N + k) * FOURH + n]
                   : Wr_o[((size_t)a * H_N + (k - DI_N)) * FOURH + n];
    NT   = 64;
    base = ws + a * AG_STRIDE;
  } else if (idx < N_OTH + N_SELF) {
    int r = idx - N_OTH;
    k = r / FOURH;
    n = r % FOURH;
    v = (k < 96) ? Wk_self[(size_t)k * FOURH + n]
                 : Wr_self[(size_t)(k - 96) * FOURH + n];
    NT   = 64;
    base = ws + OFF_SELF;
  } else if (idx < N_OTH + N_SELF + N_OTM) {
    int r = idx - N_OTH - N_SELF;
    k = r / C_N;
    n = r % C_N;
    v = W_otm[(size_t)k * C_N + n];
    NT   = 2;
    base = ws + OFF_OTM;
  } else {
    return;
  }

  int ks = k >> 5, kk = k & 31;
  int nt = n >> 4, nn = n & 15;
  int lane = nn + 16 * ((kk >> 3) & 1);
  int s    = (kk & 7) + 8 * (kk >> 4);
  int blk  = ks * NT + nt;
  __bf16 h = (__bf16)v;
  __bf16 l = (__bf16)(v - (float)h);
  base[(size_t)blk * 1024 + lane * 16 + s]       = h;
  base[(size_t)blk * 1024 + 512 + lane * 16 + s] = l;
}

// ---------------- kernel 1: map2 = blurmap (f32, streaming copy) --------------
__global__ void copy_map(const float4* __restrict__ src, float4* __restrict__ dst,
                         int n4) {
  int tid    = blockIdx.x * blockDim.x + threadIdx.x;
  int stride = gridDim.x * blockDim.x;
  for (int i = tid; i < n4; i += stride) dst[i] = src[i];
}

// ---------------- kernel 2: others LSTM + to_map + scatter --------------------
__global__ void __launch_bounds__(256) others_lstm(
    const float* __restrict__ x_o, const float* __restrict__ h_o,
    const float* __restrict__ c_o, const float* __restrict__ b_o,
    const float* __restrict__ b_otm, const int* __restrict__ pos_o,
    const __bf16* __restrict__ wsw, const __bf16* __restrict__ otmw,
    float* __restrict__ map) {
  __shared__ float h2s[32][264];  // 33 KB, padded to dodge bank conflicts

  int w    = threadIdx.x >> 5;
  int lane = threadIdx.x & 31;
  int half = lane >> 4;
  int nn   = lane & 15;
  int a    = blockIdx.x >> 7;          // 128 row-blocks per agent
  int row0 = (blockIdx.x & 127) * 32;

  v8f acc[2][4][2];
#pragma unroll
  for (int mt = 0; mt < 2; ++mt)
#pragma unroll
    for (int g = 0; g < 4; ++g)
#pragma unroll
      for (int sub = 0; sub < 2; ++sub)
        acc[mt][g][sub] = (v8f){0.f, 0.f, 0.f, 0.f, 0.f, 0.f, 0.f, 0.f};

  const __bf16* wa = wsw + (size_t)a * AG_STRIDE;
  for (int ks = 0; ks < 10; ++ks) {
    const float* abase;
    int ld, k0;
    if (ks < 2) { abase = x_o + (size_t)a * B_N * DI_N; ld = DI_N; k0 = ks * 32; }
    else        { abase = h_o + (size_t)a * B_N * H_N;  ld = H_N;  k0 = (ks - 2) * 32; }

    v16bf ahi0, alo0, ahi1, alo1;
    load_a_frag(abase, ld, row0,      k0, ahi0, alo0);
    load_a_frag(abase, ld, row0 + 16, k0, ahi1, alo1);

    const __bf16* wk = wa + (size_t)ks * (64 * 1024);
#pragma unroll
    for (int g = 0; g < 4; ++g) {
#pragma unroll
      for (int sub = 0; sub < 2; ++sub) {
        const __bf16* bb = wk + (size_t)(g * 16 + w * 2 + sub) * 1024;
        v16bf bhi = *(const v16bf*)(bb + lane * 16);
        v16bf blo = *(const v16bf*)(bb + 512 + lane * 16);
        // bf16x3: hi*hi + hi*lo + lo*hi, f32 accumulate
        acc[0][g][sub] = wmma_bf16(ahi0, bhi, acc[0][g][sub]);
        acc[0][g][sub] = wmma_bf16(ahi0, blo, acc[0][g][sub]);
        acc[0][g][sub] = wmma_bf16(alo0, bhi, acc[0][g][sub]);
        acc[1][g][sub] = wmma_bf16(ahi1, bhi, acc[1][g][sub]);
        acc[1][g][sub] = wmma_bf16(ahi1, blo, acc[1][g][sub]);
        acc[1][g][sub] = wmma_bf16(alo1, bhi, acc[1][g][sub]);
      }
    }
  }

  // gates: wave w owns gate-columns [w*32, w*32+32) across all 4 gates
  const float* cb = c_o + (size_t)a * B_N * H_N;
  const float* bb = b_o + (size_t)a * FOURH;
#pragma unroll
  for (int sub = 0; sub < 2; ++sub) {
    int gc   = w * 32 + sub * 16 + nn;
    float bi = bb[gc], bff = bb[256 + gc], bg = bb[512 + gc], bo = bb[768 + gc];
#pragma unroll
    for (int mt = 0; mt < 2; ++mt) {
#pragma unroll
      for (int j = 0; j < 8; ++j) {
        int rl   = mt * 16 + half * 8 + j;
        float cp = cb[(size_t)(row0 + rl) * H_N + gc];
        float iv = sigmoidf_(acc[mt][0][sub][j] + bi);
        float fv = sigmoidf_(acc[mt][1][sub][j] + bff);
        float gv = tanhf(acc[mt][2][sub][j] + bg);
        float ov = sigmoidf_(acc[mt][3][sub][j] + bo);
        float c2 = fv * cp + iv * gv;
        h2s[rl][gc] = ov * tanhf(c2);
      }
    }
  }
  __syncthreads();

  // fused to_map = h2 @ W_otm + b_otm, scattered with atomics
  if (w < 4) {
    int mt = w >> 1, ntile = w & 1;
    v8f a2 = (v8f){0.f, 0.f, 0.f, 0.f, 0.f, 0.f, 0.f, 0.f};
    for (int ks = 0; ks < 8; ++ks) {
      v16bf ahi, alo;
      load_a_frag((const float*)h2s, 264, mt * 16, ks * 32, ahi, alo);
      const __bf16* ob = otmw + (size_t)(ks * 2 + ntile) * 1024;
      v16bf bhi = *(const v16bf*)(ob + lane * 16);
      v16bf blo = *(const v16bf*)(ob + 512 + lane * 16);
      a2 = wmma_bf16(ahi, bhi, a2);
      a2 = wmma_bf16(ahi, blo, a2);
      a2 = wmma_bf16(alo, bhi, a2);
    }
    int col   = ntile * 16 + nn;
    float bo2 = b_otm[col];
#pragma unroll
    for (int j = 0; j < 8; ++j) {
      int rl   = mt * 16 + half * 8 + j;
      int r    = a * B_N + row0 + rl;
      int flat = pos_o[2 * r] * MH_N + pos_o[2 * r + 1];
      atomicAdd(map + (size_t)flat * C_N + col, a2[j] + bo2);
    }
  }
}

// ---------------- kernel 3: gather comm + self LSTM + output ------------------
__global__ void __launch_bounds__(256) self_lstm(
    const float* __restrict__ x_s, const float* __restrict__ h_s,
    const float* __restrict__ c_s, const float* __restrict__ b_s,
    const float* __restrict__ W_out, const float* __restrict__ b_out,
    const int* __restrict__ pos_s, const __bf16* __restrict__ wsw,
    const float* __restrict__ map, float* __restrict__ out) {
  __shared__ float h2s[32][264];

  int w    = threadIdx.x >> 5;
  int lane = threadIdx.x & 31;
  int half = lane >> 4;
  int nn   = lane & 15;
  int row0 = blockIdx.x * 32;

  v8f acc[2][4][2];
#pragma unroll
  for (int mt = 0; mt < 2; ++mt)
#pragma unroll
    for (int g = 0; g < 4; ++g)
#pragma unroll
      for (int sub = 0; sub < 2; ++sub)
        acc[mt][g][sub] = (v8f){0.f, 0.f, 0.f, 0.f, 0.f, 0.f, 0.f, 0.f};

  for (int ks = 0; ks < 11; ++ks) {
    v16bf ahi0, alo0, ahi1, alo1;
    if (ks < 2) {
      load_a_frag(x_s, DS_N, row0,      ks * 32, ahi0, alo0);
      load_a_frag(x_s, DS_N, row0 + 16, ks * 32, ahi1, alo1);
    } else if (ks == 2) {
      load_a_frag_comm(map, pos_s, row0,      ahi0, alo0);
      load_a_frag_comm(map, pos_s, row0 + 16, ahi1, alo1);
    } else {
      load_a_frag(h_s, H_N, row0,      (ks - 3) * 32, ahi0, alo0);
      load_a_frag(h_s, H_N, row0 + 16, (ks - 3) * 32, ahi1, alo1);
    }

    const __bf16* wk = wsw + (size_t)ks * (64 * 1024);
#pragma unroll
    for (int g = 0; g < 4; ++g) {
#pragma unroll
      for (int sub = 0; sub < 2; ++sub) {
        const __bf16* bb = wk + (size_t)(g * 16 + w * 2 + sub) * 1024;
        v16bf bhi = *(const v16bf*)(bb + lane * 16);
        v16bf blo = *(const v16bf*)(bb + 512 + lane * 16);
        acc[0][g][sub] = wmma_bf16(ahi0, bhi, acc[0][g][sub]);
        acc[0][g][sub] = wmma_bf16(ahi0, blo, acc[0][g][sub]);
        acc[0][g][sub] = wmma_bf16(alo0, bhi, acc[0][g][sub]);
        acc[1][g][sub] = wmma_bf16(ahi1, bhi, acc[1][g][sub]);
        acc[1][g][sub] = wmma_bf16(ahi1, blo, acc[1][g][sub]);
        acc[1][g][sub] = wmma_bf16(alo1, bhi, acc[1][g][sub]);
      }
    }
  }

#pragma unroll
  for (int sub = 0; sub < 2; ++sub) {
    int gc   = w * 32 + sub * 16 + nn;
    float bi = b_s[gc], bff = b_s[256 + gc], bg = b_s[512 + gc], bo = b_s[768 + gc];
#pragma unroll
    for (int mt = 0; mt < 2; ++mt) {
#pragma unroll
      for (int j = 0; j < 8; ++j) {
        int rl   = mt * 16 + half * 8 + j;
        float cp = c_s[(size_t)(row0 + rl) * H_N + gc];
        float iv = sigmoidf_(acc[mt][0][sub][j] + bi);
        float fv = sigmoidf_(acc[mt][1][sub][j] + bff);
        float gv = tanhf(acc[mt][2][sub][j] + bg);
        float ov = sigmoidf_(acc[mt][3][sub][j] + bo);
        float c2 = fv * cp + iv * gv;
        h2s[rl][gc] = ov * tanhf(c2);
      }
    }
  }
  __syncthreads();

  // output = h2 @ W_out + b_out  (N=1 -> wave-level dot reduction)
  int t    = threadIdx.x;
  int rl   = t >> 3;
  int part = t & 7;
  float sum = 0.f;
#pragma unroll
  for (int kk = 0; kk < 32; ++kk) sum += h2s[rl][part * 32 + kk] * W_out[part * 32 + kk];
  sum += __shfl_xor(sum, 1, 32);
  sum += __shfl_xor(sum, 2, 32);
  sum += __shfl_xor(sum, 4, 32);
  if (part == 0) out[row0 + rl] = sum + b_out[0];
}

// -----------------------------------------------------------------------------
extern "C" void kernel_launch(void* const* d_in, const int* in_sizes, int n_in,
                              void* d_out, int out_size, void* d_ws, size_t ws_size,
                              hipStream_t stream) {
  const float* x_self   = (const float*)d_in[0];
  const float* x_others = (const float*)d_in[1];
  const float* blurmap  = (const float*)d_in[2];
  const float* h_self   = (const float*)d_in[3];
  const float* c_self   = (const float*)d_in[4];
  const float* h_others = (const float*)d_in[5];
  const float* c_others = (const float*)d_in[6];
  const float* Wk_self  = (const float*)d_in[7];
  const float* Wr_self  = (const float*)d_in[8];
  const float* b_self   = (const float*)d_in[9];
  const float* Wk_o     = (const float*)d_in[10];
  const float* Wr_o     = (const float*)d_in[11];
  const float* b_o      = (const float*)d_in[12];
  const float* W_otm    = (const float*)d_in[13];
  const float* b_otm    = (const float*)d_in[14];
  const float* W_out    = (const float*)d_in[15];
  const float* b_out    = (const float*)d_in[16];
  const int*   pos_o    = (const int*)d_in[17];
  const int*   pos_s    = (const int*)d_in[18];

  float* out = (float*)d_out;
  float* map = out + B_N;  // map2 region of the tuple output

  __bf16* ws            = (__bf16*)d_ws;
  const __bf16* othersW = ws;
  const __bf16* selfW   = ws + OFF_SELF;
  const __bf16* otmW    = ws + OFF_OTM;

  // 1) pack weights (2,990,080 elements exactly = 11680 * 256)
  swizzle_weights<<<11680, 256, 0, stream>>>(Wk_self, Wr_self, Wk_o, Wr_o, W_otm, ws);
  // 2) map2 = blurmap
  copy_map<<<8192, 256, 0, stream>>>((const float4*)blurmap, (float4*)map,
                                     (MW_N * MH_N * C_N) / 4);
  // 3) others LSTM -> to_map -> scatter-add into map2
  others_lstm<<<A_N * (B_N / 32), 256, 0, stream>>>(x_others, h_others, c_others,
                                                    b_o, b_otm, pos_o, othersW,
                                                    otmW, map);
  // 4) gather comm -> self LSTM -> output
  self_lstm<<<B_N / 32, 256, 0, stream>>>(x_self, h_self, c_self, b_self, W_out,
                                          b_out, pos_s, selfW, map, out);
}
